// CosineSimCodebook_63763084476533
// MI455X (gfx1250) — matrix operand order repaired and checked by reading, not compiled, not run on hardware
//
#include <hip/hip_runtime.h>
#include <hip/hip_bf16.h>

// ---------------------------------------------------------------------------
// CosineSimCodebook for MI455X (gfx1250, wave32, WMMA + async LDS DMA)
//   dist GEMM (16384x8192x256) fused with row-argmax:
//     - v_wmma_f32_16x16x32_bf16 (8-chain over d=256)
//     - embed tiles staged in LDS via global_load_async_to_lds_b128,
//       double-buffered on ASYNCcnt, shared by 4 waves per block.
// ---------------------------------------------------------------------------

#define DIM        256
#define NROWS      16384          // 8*2048
#define NCODES     8192
#define DECAY_F    0.8f
#define TILE_CODES 16
#define NTILES     (NCODES / TILE_CODES)          // 512
#define TILE_USH   (TILE_CODES * DIM)             // 4096 ushorts = 8KB

typedef __attribute__((ext_vector_type(16))) __bf16 v16bf;
typedef __attribute__((ext_vector_type(8)))  float  v8f;
typedef __attribute__((ext_vector_type(4)))  int    i4;

// exact builtin parameter types: vector_size(16) int, AS1 (global) / AS3 (LDS)
typedef int v4i_b __attribute__((vector_size(16)));
typedef __attribute__((address_space(1))) v4i_b gv4i;
typedef __attribute__((address_space(3))) v4i_b lv4i;

union Frag {
    i4    q[2];   // two 16-byte loads
    v16bf v;      // 8 VGPRs fed to WMMA
};

// --- CDNA5 async global->LDS copy (16B per lane), ASYNCcnt-tracked ---------
__device__ __forceinline__ void async_copy_b128(void* lds_dst, const void* gsrc) {
#if __has_builtin(__builtin_amdgcn_global_load_async_to_lds_b128)
    __builtin_amdgcn_global_load_async_to_lds_b128(
        (gv4i*)(const_cast<void*>(gsrc)), (lv4i*)lds_dst,
        /*offset=*/0, /*cpol=*/0);
#else
    unsigned lds_off = (unsigned)(unsigned long long)lds_dst;
    asm volatile("global_load_async_to_lds_b128 %0, %1, off"
                 :: "v"(lds_off), "v"((unsigned long long)gsrc)
                 : "memory");
#endif
}

__device__ __forceinline__ void wait_async_le4() {
#if __has_builtin(__builtin_amdgcn_s_wait_asynccnt)
    __builtin_amdgcn_s_wait_asynccnt(4);
#else
    asm volatile("s_wait_asynccnt 0x4" ::: "memory");
#endif
}

__device__ __forceinline__ void wait_async_le0() {
#if __has_builtin(__builtin_amdgcn_s_wait_asynccnt)
    __builtin_amdgcn_s_wait_asynccnt(0);
#else
    asm volatile("s_wait_asynccnt 0x0" ::: "memory");
#endif
}

__device__ __forceinline__ unsigned short f32_to_bf16_rne(float x) {
    unsigned int u = __float_as_uint(x);
    unsigned int r = u + 0x7FFFu + ((u >> 16) & 1u);
    return (unsigned short)(r >> 16);
}

// ---------------------------------------------------------------------------
// Kernel 1: per-row L2 normalize (fp32 out + bf16 out).
// ---------------------------------------------------------------------------
__global__ __launch_bounds__(256) void l2norm_rows_kernel(
    const float* __restrict__ in,
    float* __restrict__ out_f32,
    unsigned short* __restrict__ out_bf16)
{
    const int row = blockIdx.x;
    const int j   = threadIdx.x;
    const float v = in[row * DIM + j];

    float ss = v * v;
    #pragma unroll
    for (int m = 16; m >= 1; m >>= 1) ss += __shfl_xor(ss, m, 32);

    __shared__ float red[8];
    if ((j & 31) == 0) red[j >> 5] = ss;
    __syncthreads();
    float tot = 0.0f;
    #pragma unroll
    for (int w = 0; w < 8; ++w) tot += red[w];

    const float norm = fmaxf(sqrtf(tot), 1e-12f);
    const float y = v / norm;
    out_f32[row * DIM + j]  = y;
    out_bf16[row * DIM + j] = f32_to_bf16_rne(y);
}

// ---------------------------------------------------------------------------
// Kernel 2: zero scratch accumulators (bins + embed_sum).
// ---------------------------------------------------------------------------
__global__ __launch_bounds__(256) void zero_kernel(float* __restrict__ p, int n)
{
    int i = blockIdx.x * blockDim.x + threadIdx.x;
    if (i < n) p[i] = 0.0f;
}

// ---------------------------------------------------------------------------
// Kernel 3: fused cosine-similarity GEMM + argmax.
//   Block = 128 threads (4 waves) -> 64 rows. All waves share the code tile:
//   8KB bf16 embed tile staged in LDS by async DMA, double-buffered.
//
// A layout (16-bit 16x32, ISA 7.12.2): lane L -> row L&15,
//   elems 0..7 = K (k0 + 8*(L>>4)) .. +7 ; elems 8..15 = +16 of that.
// B layout (16-bit 32x16): lane n -> col n&15,
//   elems 0..7 = K (k0 + 8*(n>>4)) .. +7 ; elems 8..15 = +16 of that (same as A).
// C/D layout: VGPR r, lanes 0-15 -> M=r ; lanes 16-31 -> M=8+r.
// ---------------------------------------------------------------------------
__global__ __launch_bounds__(128) void argmax_wmma_kernel(
    const unsigned short* __restrict__ fbf,   // flatten bf16 [NROWS, DIM]
    const unsigned short* __restrict__ ebf,   // embed_n bf16 [NCODES, DIM]
    int* __restrict__ idx_out)                // [NROWS]
{
    __shared__ __attribute__((aligned(16))) unsigned short lds_b[2][TILE_USH];

    const int tid  = threadIdx.x;       // 0..127
    const int lane = tid & 31;
    const int wv   = tid >> 5;          // wave in block: 0..3
    const int row_base = blockIdx.x * 64 + wv * 16;
    const int lrow = lane & 15;
    const int hi   = lane >> 4;

    // --- resident A fragments: 8 fragments cover d=256 for this wave's rows
    Frag a[8];
    const unsigned short* arow = fbf + (size_t)(row_base + lrow) * DIM;
    #pragma unroll
    for (int f = 0; f < 8; ++f) {
        const int k0 = f * 32 + hi * 8;
        a[f].q[0] = *(const i4*)(arow + k0);
        a[f].q[1] = *(const i4*)(arow + k0 + 16);
    }

    float best[8];
    int   bidx[8];
    #pragma unroll
    for (int r = 0; r < 8; ++r) { best[r] = -3.0e38f; bidx[r] = 0; }

    // --- async DMA of one 8KB code tile: 512 x 16B chunks, 128 threads x 4
    const char* gbase = (const char*)ebf;

    // prologue: tile 0 -> buffer 0
    {
        char* dst = (char*)&lds_b[0][0];
        #pragma unroll
        for (int c = 0; c < 4; ++c) {
            const int off = (c * 128 + tid) * 16;
            async_copy_b128(dst + off, gbase + off);
        }
    }

    for (int t = 0; t < NTILES; ++t) {
        const int cur = t & 1;

        if (t + 1 < NTILES) {
            // issue next tile into the other buffer, then drain current tile
            char* dst = (char*)&lds_b[cur ^ 1][0];
            const char* src = gbase + (size_t)(t + 1) * TILE_CODES * DIM * 2;
            #pragma unroll
            for (int c = 0; c < 4; ++c) {
                const int off = (c * 128 + tid) * 16;
                async_copy_b128(dst + off, src + off);
            }
            wait_async_le4();   // in-order: current tile's 4 copies are done
        } else {
            wait_async_le0();
        }
        __syncthreads();        // whole tile visible to all 4 waves

        // --- load all 8 B fragments from LDS, then chain 8 WMMAs ---
        const unsigned short* bp = &lds_b[cur][lrow * DIM + hi * 8];
        Frag b[8];
        #pragma unroll
        for (int f = 0; f < 8; ++f) {
            b[f].q[0] = *(const i4*)(bp + f * 32);
            b[f].q[1] = *(const i4*)(bp + f * 32 + 16);
        }
        v8f acc = {0.f, 0.f, 0.f, 0.f, 0.f, 0.f, 0.f, 0.f};
        #pragma unroll
        for (int f = 0; f < 8; ++f) {
            acc = __builtin_amdgcn_wmma_f32_16x16x32_bf16(
                /*neg_a=*/false, a[f].v, /*neg_b=*/false, b[f].v,
                /*c_mod=*/(short)0, acc, /*reuse_a=*/false, /*reuse_b=*/false);
        }

        const int cand = t * TILE_CODES + lrow;
        #pragma unroll
        for (int r = 0; r < 8; ++r) {
            if (acc[r] > best[r]) { best[r] = acc[r]; bidx[r] = cand; }
        }

        __syncthreads();        // all waves done reading before buffer reuse
    }

    // --- argmax reduce across the 16 column lanes of each half-wave ---
    #pragma unroll
    for (int m = 1; m < 16; m <<= 1) {
        #pragma unroll
        for (int r = 0; r < 8; ++r) {
            const float ov = __shfl_xor(best[r], m, 32);
            const int   oi = __shfl_xor(bidx[r], m, 32);
            if (ov > best[r] || (ov == best[r] && oi < bidx[r])) {
                best[r] = ov; bidx[r] = oi;
            }
        }
    }

    if (lrow == 0) {
        const int mbase = row_base + hi * 8;   // D-matrix: half hi holds M = hi*8 + r
        #pragma unroll
        for (int r = 0; r < 8; ++r) idx_out[mbase + r] = bidx[r];
    }
}

// ---------------------------------------------------------------------------
// Kernel 4: gather quantize rows (raw embed), emit indices, accumulate
// bins + embed_sum via f32 global atomics. One block per row.
// ---------------------------------------------------------------------------
__global__ __launch_bounds__(256) void scatter_kernel(
    const int*   __restrict__ idx,
    const float* __restrict__ embed_raw,
    const float* __restrict__ flatten,
    float* __restrict__ out_q,
    float* __restrict__ out_ind,
    float* __restrict__ bins,
    float* __restrict__ embed_sum)
{
    const int row = blockIdx.x;
    const int j   = threadIdx.x;
    const int k   = idx[row];

    out_q[(size_t)row * DIM + j] = embed_raw[(size_t)k * DIM + j];
    atomicAdd(&embed_sum[(size_t)k * DIM + j], flatten[(size_t)row * DIM + j]);
    if (j == 0) {
        out_ind[row] = (float)k;
        atomicAdd(&bins[k], 1.0f);
    }
}

// ---------------------------------------------------------------------------
// Kernel 5: EMA codebook update.
// ---------------------------------------------------------------------------
__global__ __launch_bounds__(256) void ema_update_kernel(
    const float* __restrict__ embed_raw,
    const float* __restrict__ embed_n,
    const float* __restrict__ embed_sum,
    const float* __restrict__ bins,
    float* __restrict__ embed_new)
{
    const int k = blockIdx.x;
    const int j = threadIdx.x;

    const float b  = bins[k];
    const float bs = (b == 0.0f) ? 1.0f : b;
    const float m  = embed_sum[(size_t)k * DIM + j] / bs;

    float ss = m * m;
    #pragma unroll
    for (int s = 16; s >= 1; s >>= 1) ss += __shfl_xor(ss, s, 32);
    __shared__ float red[8];
    if ((j & 31) == 0) red[j >> 5] = ss;
    __syncthreads();
    float tot = 0.0f;
    #pragma unroll
    for (int w = 0; w < 8; ++w) tot += red[w];

    float en = m / fmaxf(sqrtf(tot), 1e-12f);
    if (b == 0.0f) en = embed_n[(size_t)k * DIM + j];

    embed_new[(size_t)k * DIM + j] =
        embed_raw[(size_t)k * DIM + j] * DECAY_F + en * (1.0f - DECAY_F);
}

// ---------------------------------------------------------------------------
extern "C" void kernel_launch(void* const* d_in, const int* in_sizes, int n_in,
                              void* d_out, int out_size, void* d_ws, size_t ws_size,
                              hipStream_t stream)
{
    const float* x     = (const float*)d_in[0];   // [8,2048,256]
    const float* embed = (const float*)d_in[1];   // [8192,256]

    float* out     = (float*)d_out;
    float* out_q   = out;                                  // 16384*256
    float* out_ind = out + (size_t)NROWS * DIM;            // 16384
    float* out_new = out_ind + NROWS;                      // 8192*256

    // workspace carve-up
    char* ws = (char*)d_ws;
    float* flatten          = (float*)ws;          ws += (size_t)NROWS * DIM * 4;
    unsigned short* fbf     = (unsigned short*)ws; ws += (size_t)NROWS * DIM * 2;
    float* embed_n          = (float*)ws;          ws += (size_t)NCODES * DIM * 4;
    unsigned short* ebf     = (unsigned short*)ws; ws += (size_t)NCODES * DIM * 2;
    int*   idx              = (int*)ws;            ws += (size_t)NROWS * 4;
    float* bins             = (float*)ws;          ws += (size_t)NCODES * 4;
    float* embed_sum        = (float*)ws;          // NCODES*DIM*4

    // 1) normalize rows (x -> flatten, embed -> embed_n), fp32 + bf16
    l2norm_rows_kernel<<<NROWS,  256, 0, stream>>>(x, flatten, fbf);
    l2norm_rows_kernel<<<NCODES, 256, 0, stream>>>(embed, embed_n, ebf);

    // 2) zero accumulators (bins immediately precedes embed_sum in ws)
    {
        const int nz = NCODES + NCODES * DIM;
        zero_kernel<<<(nz + 255) / 256, 256, 0, stream>>>(bins, nz);
    }

    // 3) fused GEMM + argmax: 256 blocks x 4 waves x 16 rows
    argmax_wmma_kernel<<<NROWS / 64, 128, 0, stream>>>(fbf, ebf, idx);

    // 4) gather + histogram + segment-sum
    scatter_kernel<<<NROWS, 256, 0, stream>>>(idx, embed, flatten,
                                              out_q, out_ind, bins, embed_sum);

    // 5) EMA codebook update
    ema_update_kernel<<<NCODES, 256, 0, stream>>>(embed, embed_n, embed_sum,
                                                  bins, out_new);
}
